// TemporalRGCN_19885698581030
// MI455X (gfx1250) — compile-verified, matrix operand order checked by reading, not run on hardware
//
#include <hip/hip_runtime.h>
#include <hip/hip_bf16.h>

// ---------------- problem constants (match reference) ----------------
#define Nn   131072
#define Ee   2097152
#define Ff   32
#define Hh   128
#define OUTn 8
#define Rr   10
#define Bb   4
#define Ll   3

// meta[] int layout in workspace
#define MI_RELCNT 0   // R
#define MI_RELOFF 16  // R+1
#define MI_CURSOR 32  // R
#define MI_SUPOFF 48  // R+1  (supertile = 256 edges)

typedef __attribute__((ext_vector_type(16))) __bf16 v16bf;
typedef __attribute__((ext_vector_type(8)))  __bf16 v8bf;
typedef __attribute__((ext_vector_type(8)))  float  v8f;

union V16U { v16bf v; v8bf h[2]; };

__device__ inline __bf16 to_bf(float f) { return (__bf16)f; }

// A/B fragment loader: 16 bf16 per lane as two 16B loads.
// Per ISA layout for 16-bit 16x32 A (and mirrored B): lane half selects +8 K
// offset (folded into the pointer by the caller); elements 0..7 = K base+0..7,
// elements 8..15 = K base+16..23.
__device__ inline v16bf ld_frag16(const __bf16* p) {
  V16U u;
  u.h[0] = *(const v8bf*)(p);
  u.h[1] = *(const v8bf*)(p + 16);
  return u.v;
}

__device__ inline v8f wmma_bf16(v16bf a, v16bf b, v8f c) {
  // (neg_a, A, neg_b, B, c_mod, C, reuse_a, reuse_b)
  return __builtin_amdgcn_wmma_f32_16x16x32_bf16(false, a, false, b, (short)0, c,
                                                 false, false);
}

// LDS W^T strides (halves). Rows must be 16B-aligned for b128 ds/async ops and
// spread the 16 column-lanes of a B-fragment over disjoint bank groups.
//  SW  = 136 halves (272B = 17*16B): lanes hit 16 disjoint 4-bank groups.
//  SW2 =  56 halves (112B =  7*16B): 28n mod 64 distinct for n=0..15.
#define SW  136
#define SW2 56

// -------- async global->LDS staging (gfx1250 ASYNCcnt path), with fallback ----
#if defined(__has_builtin)
#if __has_builtin(__builtin_amdgcn_global_load_async_to_lds_b128) && \
    __has_builtin(__builtin_amdgcn_s_wait_asynccnt)
#define HAVE_ASYNC_LDS 1
#endif
#endif

#if defined(HAVE_ASYNC_LDS)
// Builtin expects pointers to 16B int vectors (per compiler diagnostic):
//   arg0: global source (AS1), arg1: LDS dest (AS3), arg2: imm offset, arg3: cpol
typedef int v4i_ __attribute__((vector_size(16)));
typedef __attribute__((address_space(1))) v4i_ g_v4i;
typedef __attribute__((address_space(3))) v4i_ l_v4i;

__device__ inline void async_cp16(__bf16* lds_dst, const __bf16* gsrc) {
  __builtin_amdgcn_global_load_async_to_lds_b128((g_v4i*)gsrc, (l_v4i*)lds_dst,
                                                 0, 0);
}
#endif

// Stage W^T (already transposed in global: [NROWS][KC], KC halves per row,
// contiguous) into LDS with row stride LSTRIDE halves. 16B chunks only.
template <int NROWS, int KC, int LSTRIDE>
__device__ inline void stage_wt(const __bf16* g, __bf16* s, int tid) {
  constexpr int CPR = KC / 8;            // 16B chunks per row
  constexpr int CH = NROWS * CPR;        // total chunks
  for (int c = tid; c < CH; c += 128) {
    int n = c / CPR, q = c - n * CPR;
#if defined(HAVE_ASYNC_LDS)
    async_cp16(s + n * LSTRIDE + q * 8, g + n * KC + q * 8);
#else
    *(v8bf*)(s + n * LSTRIDE + q * 8) = *(const v8bf*)(g + n * KC + q * 8);
#endif
  }
#if defined(HAVE_ASYNC_LDS)
  __builtin_amdgcn_s_wait_asynccnt(0);   // wave-local copies done; barrier next
#endif
}

// ---------------- small utility kernels ----------------
__global__ void k_zero_f32(float* p, int n) {
  int i = blockIdx.x * blockDim.x + threadIdx.x;
  if (i < n) p[i] = 0.0f;
}

__global__ void k_cvt_bf16(const float* s, __bf16* d, int n) {
  int i = blockIdx.x * blockDim.x + threadIdx.x;
  if (i < n) d[i] = to_bf(s[i]);
}

// transposed convert for L batches of [K][N] f32 -> [N][K] bf16
__global__ void k_cvt_tr(const float* s, __bf16* d, int K, int N, int total) {
  int idx = blockIdx.x * blockDim.x + threadIdx.x;
  if (idx >= total) return;
  int kn = K * N;
  int l = idx / kn, rem = idx - l * kn;
  int n = rem / K, k = rem - n * K;
  d[idx] = to_bf(s[l * kn + k * N + n]);
}

__global__ void k_count(const int* ei, const int* et, float* cnt, int* meta) {
  int e = blockIdx.x * blockDim.x + threadIdx.x;
  if (e >= Ee) return;
  int d = ei[Ee + e];
  int r = et[e];
  atomicAdd(&cnt[(size_t)d * Rr + r], 1.0f);
  atomicAdd(&meta[MI_RELCNT + r], 1);
}

__global__ void k_sched(int* meta) {
  if (threadIdx.x != 0) return;
  int off = 0;
  for (int r = 0; r < Rr; ++r) {
    meta[MI_RELOFF + r] = off;
    meta[MI_CURSOR + r] = off;
    off += meta[MI_RELCNT + r];
  }
  meta[MI_RELOFF + Rr] = off;
  int so = 0;
  for (int r = 0; r < Rr; ++r) {
    meta[MI_SUPOFF + r] = so;
    so += (meta[MI_RELCNT + r] + 255) >> 8;  // 256-edge supertiles
  }
  meta[MI_SUPOFF + Rr] = so;
}

__global__ void k_bin(const int* et, int* meta, int* perm) {
  int e = blockIdx.x * blockDim.x + threadIdx.x;
  if (e >= Ee) return;
  int r = et[e];
  int pos = atomicAdd(&meta[MI_CURSOR + r], 1);
  perm[pos] = e;
}

// W_r^T = (sum_b comp[l,r,b] * basis[l,b])^T  ->  bf16 [L,R,n,k] (983 KB)
__global__ void k_prep_wrel(const float* basis, const float* comp,
                            __bf16* wrelT) {
  int idx = blockIdx.x * blockDim.x + threadIdx.x;
  if (idx >= Ll * Rr * Hh * Hh) return;
  int k = idx & 127;
  int n = (idx >> 7) & 127;
  int rl = idx >> 14;           // H*H == 1<<14
  int l = rl / Rr, r = rl - l * Rr;
  float a = 0.0f;
#pragma unroll
  for (int b = 0; b < Bb; ++b)
    a += comp[(l * Rr + r) * Bb + b] * basis[((l * Bb + b) << 14) + k * Hh + n];
  wrelT[idx] = to_bf(a);
}

// W2 padded to 16 output cols, stored pre-transposed [n=16][k=128]
__global__ void k_prep_w2p(const float* W2, __bf16* w2p) {
  int idx = blockIdx.x * blockDim.x + threadIdx.x;
  if (idx >= 16 * Hh) return;
  int nn = idx >> 7, k = idx & 127;
  w2p[idx] = to_bf(nn < OUTn ? W2[k * OUTn + nn] : 0.0f);
}

// ---------------- input projection: h = x @ W_in + b_in ----------------
__global__ __launch_bounds__(128) void k_inproj(const __bf16* xbf,
                                                const __bf16* winT,
                                                const float* b_in, float* h,
                                                __bf16* hbf) {
  __shared__ __align__(16) __bf16 sWT[128 * SW2];  // W_in^T [n=128][k=32]
  int tid = threadIdx.x;
  stage_wt<128, 32, SW2>(winT, sWT, tid);
  __syncthreads();
  int lane = tid & 31, wave = tid >> 5;
  int mm = lane & 15, off = (lane & 16) ? 8 : 0, n0a = wave * 32;
  size_t row0 = (size_t)blockIdx.x * 16;
  v16bf a = ld_frag16(xbf + (row0 + mm) * Ff + off);
  v8f acc0 = {0, 0, 0, 0, 0, 0, 0, 0}, acc1 = {0, 0, 0, 0, 0, 0, 0, 0};
  v16bf b0 = ld_frag16(sWT + (size_t)(n0a + mm) * SW2 + off);
  acc0 = wmma_bf16(a, b0, acc0);
  v16bf b1 = ld_frag16(sWT + (size_t)(n0a + 16 + mm) * SW2 + off);
  acc1 = wmma_bf16(a, b1, acc1);
  int col = n0a + mm;
#pragma unroll
  for (int r2 = 0; r2 < 8; ++r2) {
    int mr = r2 + ((lane & 16) ? 8 : 0);
    size_t base = (row0 + mr) * Hh;
    float v0 = acc0[r2] + b_in[col];
    float v1 = acc1[r2] + b_in[col + 16];
    h[base + col] = v0;        hbf[base + col] = to_bf(v0);
    h[base + col + 16] = v1;   hbf[base + col + 16] = to_bf(v1);
  }
}

// ------- relation-binned edge conv: agg[dst] += inv * h_bf[src] @ W_r -------
// One block = 256 edges of ONE relation (16 WMMA tiles), W_r staged once.
__global__ __launch_bounds__(128) void k_edge(const int* perm, const int* ei,
                                              const float* cnt, const int* meta,
                                              const __bf16* wrelT_l,
                                              const __bf16* hbf, float* agg) {
  __shared__ __align__(16) __bf16 sWT[128 * SW];
  __shared__ int s_src[16];
  __shared__ int s_dst[16];
  __shared__ float s_inv[16];

  int sid = blockIdx.x;
  int rel = -1;
#pragma unroll
  for (int r = 0; r < Rr; ++r)
    if (sid >= meta[MI_SUPOFF + r] && sid < meta[MI_SUPOFF + r + 1]) rel = r;
  if (rel < 0) return;
  int ebase = meta[MI_RELOFF + rel] + (sid - meta[MI_SUPOFF + rel]) * 256;
  int eend = meta[MI_RELOFF + rel + 1];

  int tid = threadIdx.x;
  stage_wt<128, 128, SW>(wrelT_l + (size_t)rel * Hh * Hh, sWT, tid);

  int lane = tid & 31, wave = tid >> 5;
  int mm = lane & 15, off = (lane & 16) ? 8 : 0, n0a = wave * 32;

  for (int t = 0; t < 16; ++t) {
    int estart = ebase + t * 16;
    if (estart >= eend) break;  // uniform across block
    int nv = eend - estart; if (nv > 16) nv = 16;
    __syncthreads();  // W stage done (t==0) / previous tile's reads done
    if (tid < 16) {
      int i = tid < nv ? tid : (nv - 1);
      int e = perm[estart + i];
      int s = ei[e], d = ei[Ee + e];
      s_src[tid] = s;
      s_dst[tid] = d;
      float c = cnt[(size_t)d * Rr + rel];
      s_inv[tid] = (tid < nv) ? (1.0f / fmaxf(c, 1.0f)) : 0.0f;  // pad lanes: 0
      __builtin_prefetch(hbf + (size_t)s * Hh, 0, 0);  // global_prefetch_b8
    }
    __syncthreads();

    const __bf16* arow = hbf + (size_t)s_src[mm] * Hh + off;  // gathered rows
    v8f acc0 = {0, 0, 0, 0, 0, 0, 0, 0}, acc1 = {0, 0, 0, 0, 0, 0, 0, 0};
#pragma unroll
    for (int kt = 0; kt < 4; ++kt) {
      v16bf a = ld_frag16(arow + kt * 32);
      int kb = kt * 32 + off;
      v16bf b0 = ld_frag16(sWT + (size_t)(n0a + mm) * SW + kb);
      acc0 = wmma_bf16(a, b0, acc0);
      v16bf b1 = ld_frag16(sWT + (size_t)(n0a + 16 + mm) * SW + kb);
      acc1 = wmma_bf16(a, b1, acc1);
    }
    int col = n0a + mm;
#pragma unroll
    for (int r2 = 0; r2 < 8; ++r2) {
      int mr = r2 + ((lane & 16) ? 8 : 0);
      float inv = s_inv[mr];
      size_t base = (size_t)s_dst[mr] * Hh;
      atomicAdd(&agg[base + col], acc0[r2] * inv);        // lands in L2
      atomicAdd(&agg[base + col + 16], acc1[r2] * inv);
    }
  }
}

// --- fused: h_new = LN(agg + h@root + bias); relu; h = (resid? h:0) + h_new ---
__global__ __launch_bounds__(128) void k_combine(const float* agg,
                                                 const __bf16* rootT_l,
                                                 const float* bias_l,
                                                 const float* g_l,
                                                 const float* bl_l, float* h,
                                                 __bf16* hbf, int resid) {
  __shared__ __align__(16) __bf16 sWT[128 * SW];
  __shared__ float s_out[16 * 132];
  __shared__ float2 s_red[16][8];
  __shared__ float2 s_stat[16];
  int tid = threadIdx.x;
  stage_wt<128, 128, SW>(rootT_l, sWT, tid);
  __syncthreads();

  int lane = tid & 31, wave = tid >> 5;
  int mm = lane & 15, off = (lane & 16) ? 8 : 0, n0a = wave * 32;
  size_t row0 = (size_t)blockIdx.x * 16;
  const __bf16* arow = hbf + (row0 + mm) * Hh + off;
  v8f acc0 = {0, 0, 0, 0, 0, 0, 0, 0}, acc1 = {0, 0, 0, 0, 0, 0, 0, 0};
#pragma unroll
  for (int kt = 0; kt < 4; ++kt) {
    v16bf a = ld_frag16(arow + kt * 32);
    int kb = kt * 32 + off;
    v16bf b0 = ld_frag16(sWT + (size_t)(n0a + mm) * SW + kb);
    acc0 = wmma_bf16(a, b0, acc0);
    v16bf b1 = ld_frag16(sWT + (size_t)(n0a + 16 + mm) * SW + kb);
    acc1 = wmma_bf16(a, b1, acc1);
  }
  int col = n0a + mm;
#pragma unroll
  for (int r2 = 0; r2 < 8; ++r2) {
    int mr = r2 + ((lane & 16) ? 8 : 0);
    s_out[mr * 132 + col] = acc0[r2];
    s_out[mr * 132 + col + 16] = acc1[r2];
  }
  __syncthreads();

  int m = tid >> 3, j = tid & 7;  // 8 threads per row
  float sum = 0.0f, ss = 0.0f;
#pragma unroll
  for (int q = 0; q < 16; ++q) {
    int c = j * 16 + q;
    float v = agg[(row0 + m) * Hh + c] + s_out[m * 132 + c] + bias_l[c];
    s_out[m * 132 + c] = v;
    sum += v; ss += v * v;
  }
  s_red[m][j] = make_float2(sum, ss);
  __syncthreads();
  if (j == 0) {
    float S = 0.0f, S2 = 0.0f;
#pragma unroll
    for (int q = 0; q < 8; ++q) { S += s_red[m][q].x; S2 += s_red[m][q].y; }
    float mu = S * (1.0f / Hh);
    float var = S2 * (1.0f / Hh) - mu * mu;
    s_stat[m] = make_float2(mu, rsqrtf(var + 1e-5f));
  }
  __syncthreads();
  float mu = s_stat[m].x, ri = s_stat[m].y;
#pragma unroll
  for (int q = 0; q < 16; ++q) {
    int c = j * 16 + q;
    float y = (s_out[m * 132 + c] - mu) * ri * g_l[c] + bl_l[c];
    y = fmaxf(y, 0.0f);
    size_t gi = (row0 + m) * Hh + c;
    float hn = y + (resid ? h[gi] : 0.0f);
    h[gi] = hn;
    hbf[gi] = to_bf(hn);
  }
}

// ------- head: out = relu(h@W1+b1) @ W2pad + b2 (two chained WMMA GEMMs) -------
__global__ __launch_bounds__(128) void k_head(const __bf16* hbf,
                                              const __bf16* w1T,
                                              const float* b1,
                                              const __bf16* w2p,
                                              const float* b2, float* out) {
  __shared__ __align__(16) __bf16 sW1T[128 * SW];
  __shared__ __align__(16) __bf16 sW2T[16 * SW];
  __shared__ __align__(16) __bf16 sT[16 * SW];
  int tid = threadIdx.x;
  stage_wt<128, 128, SW>(w1T, sW1T, tid);
  stage_wt<16, 128, SW>(w2p, sW2T, tid);
  __syncthreads();

  int lane = tid & 31, wave = tid >> 5;
  int mm = lane & 15, off = (lane & 16) ? 8 : 0, n0a = wave * 32;
  size_t row0 = (size_t)blockIdx.x * 16;
  const __bf16* arow = hbf + (row0 + mm) * Hh + off;
  v8f acc0 = {0, 0, 0, 0, 0, 0, 0, 0}, acc1 = {0, 0, 0, 0, 0, 0, 0, 0};
#pragma unroll
  for (int kt = 0; kt < 4; ++kt) {
    v16bf a = ld_frag16(arow + kt * 32);
    int kb = kt * 32 + off;
    v16bf b0 = ld_frag16(sW1T + (size_t)(n0a + mm) * SW + kb);
    acc0 = wmma_bf16(a, b0, acc0);
    v16bf b1 = ld_frag16(sW1T + (size_t)(n0a + 16 + mm) * SW + kb);
    acc1 = wmma_bf16(a, b1, acc1);
  }
  int col = n0a + mm;
#pragma unroll
  for (int r2 = 0; r2 < 8; ++r2) {
    int mr = r2 + ((lane & 16) ? 8 : 0);
    sT[mr * SW + col] = to_bf(fmaxf(acc0[r2] + b1[col], 0.0f));
    sT[mr * SW + col + 16] = to_bf(fmaxf(acc1[r2] + b1[col + 16], 0.0f));
  }
  __syncthreads();
  if (wave == 0) {
    v8f acc = {0, 0, 0, 0, 0, 0, 0, 0};
#pragma unroll
    for (int kt = 0; kt < 4; ++kt) {
      int kb = kt * 32 + off;
      v16bf a = ld_frag16(sT + (size_t)mm * SW + kb);
      v16bf b = ld_frag16(sW2T + (size_t)mm * SW + kb);
      acc = wmma_bf16(a, b, acc);
    }
    if (mm < OUTn) {
#pragma unroll
      for (int r2 = 0; r2 < 8; ++r2) {
        int mr = r2 + ((lane & 16) ? 8 : 0);
        out[(row0 + mr) * OUTn + mm] = acc[r2] + b2[mm];
      }
    }
  }
}

// ---------------- launcher ----------------
extern "C" void kernel_launch(void* const* d_in, const int* in_sizes, int n_in,
                              void* d_out, int out_size, void* d_ws,
                              size_t ws_size, hipStream_t stream) {
  (void)in_sizes; (void)n_in; (void)out_size; (void)ws_size;
  const float* x     = (const float*)d_in[0];
  const int*   ei    = (const int*)d_in[1];
  const int*   et    = (const int*)d_in[2];
  const float* Win   = (const float*)d_in[3];
  const float* b_in  = (const float*)d_in[4];
  const float* basis = (const float*)d_in[5];
  const float* comp  = (const float*)d_in[6];
  const float* root  = (const float*)d_in[7];
  const float* bias  = (const float*)d_in[8];
  const float* ln_g  = (const float*)d_in[9];
  const float* ln_b  = (const float*)d_in[10];
  const float* W1    = (const float*)d_in[11];
  const float* b1    = (const float*)d_in[12];
  const float* W2    = (const float*)d_in[13];
  const float* b2    = (const float*)d_in[14];
  float* out = (float*)d_out;

  char* ws = (char*)d_ws;
  size_t o = 0;
  auto take = [&](size_t bytes) -> char* {
    char* p = ws + o;
    o += (bytes + 255) & ~(size_t)255;
    return p;
  };
  float*  cnt    = (float*)take((size_t)Nn * Rr * 4);        // 5.2 MB
  int*    meta   = (int*)take(64 * 4);
  int*    perm   = (int*)take((size_t)Ee * 4);               // 8.4 MB
  __bf16* xbf    = (__bf16*)take((size_t)Nn * Ff * 2);       // 8.4 MB
  __bf16* winT   = (__bf16*)take((size_t)Ff * Hh * 2);       // [n=128][k=32]
  __bf16* w1T    = (__bf16*)take((size_t)Hh * Hh * 2);       // [n][k]
  __bf16* w2p    = (__bf16*)take((size_t)16 * Hh * 2);       // [n=16][k=128]
  __bf16* rootT  = (__bf16*)take((size_t)Ll * Hh * Hh * 2);  // [l][n][k]
  __bf16* wrelT  = (__bf16*)take((size_t)Ll * Rr * Hh * Hh * 2);  // 983 KB
  float*  h      = (float*)take((size_t)Nn * Hh * 4);        // 67 MB
  __bf16* hbf    = (__bf16*)take((size_t)Nn * Hh * 2);       // 33 MB
  float*  agg    = (float*)take((size_t)Nn * Hh * 4);        // 67 MB

  k_zero_f32<<<(Nn * Rr + 255) / 256, 256, 0, stream>>>(cnt, Nn * Rr);
  k_zero_f32<<<1, 64, 0, stream>>>((float*)meta, 64);
  k_count<<<Ee / 256, 256, 0, stream>>>(ei, et, cnt, meta);
  k_sched<<<1, 32, 0, stream>>>(meta);
  k_bin<<<Ee / 256, 256, 0, stream>>>(et, meta, perm);

  k_cvt_bf16<<<(Nn * Ff + 255) / 256, 256, 0, stream>>>(x, xbf, Nn * Ff);
  k_cvt_tr<<<(Ff * Hh + 255) / 256, 256, 0, stream>>>(Win, winT, Ff, Hh,
                                                      Ff * Hh);
  k_cvt_tr<<<(Hh * Hh + 255) / 256, 256, 0, stream>>>(W1, w1T, Hh, Hh, Hh * Hh);
  k_cvt_tr<<<(Ll * Hh * Hh + 255) / 256, 256, 0, stream>>>(root, rootT, Hh, Hh,
                                                           Ll * Hh * Hh);
  k_prep_w2p<<<(16 * Hh + 255) / 256, 256, 0, stream>>>(W2, w2p);
  k_prep_wrel<<<(Ll * Rr * Hh * Hh + 255) / 256, 256, 0, stream>>>(basis, comp,
                                                                   wrelT);

  k_inproj<<<Nn / 16, 128, 0, stream>>>(xbf, winT, b_in, h, hbf);

  for (int l = 0; l < Ll; ++l) {
    k_zero_f32<<<(Nn * Hh + 255) / 256, 256, 0, stream>>>(agg, Nn * Hh);
    k_edge<<<Ee / 256 + Rr, 128, 0, stream>>>(
        perm, ei, cnt, meta, wrelT + (size_t)l * Rr * Hh * Hh, hbf, agg);
    k_combine<<<Nn / 16, 128, 0, stream>>>(
        agg, rootT + (size_t)l * Hh * Hh, bias + l * Hh, ln_g + l * Hh,
        ln_b + l * Hh, h, hbf, l > 0 ? 1 : 0);
  }

  k_head<<<Nn / 16, 128, 0, stream>>>(hbf, w1T, b1, w2p, b2, out);
}
